// Attention_43834436223075
// MI455X (gfx1250) — matrix operand order repaired and checked by reading, not compiled, not run on hardware
//
#include <hip/hip_runtime.h>

typedef _Float16 f16;
typedef __attribute__((ext_vector_type(16))) _Float16 v16h;
typedef __attribute__((ext_vector_type(8)))  _Float16 v8h;
typedef __attribute__((ext_vector_type(8)))  float    v8f;

#define BDIM 16
#define TDIM 2048
#define DDIM 128
#define QTILE 128              // 8 waves x 16 query rows
#define KTILE 64               // key tile
#define NKT   (TDIM / KTILE)   // 32
#define KPAD  (KTILE + 8)

// ---- DPP ROW_XMASK butterfly (pure VALU cross-lane, masks 1..8 stay in 16-lane halves) ----
template<int CTRL>
__device__ __forceinline__ float dpp_f(float v) {
  int i = __builtin_bit_cast(int, v);
  int r = __builtin_amdgcn_update_dpp(i, i, CTRL, 0xf, 0xf, false);
  return __builtin_bit_cast(float, r);
}
__device__ __forceinline__ void redmax8(float v[8]) {
#pragma unroll
  for (int r = 0; r < 8; ++r) v[r] = fmaxf(v[r], dpp_f<0x161>(v[r]));  // xor 1
#pragma unroll
  for (int r = 0; r < 8; ++r) v[r] = fmaxf(v[r], dpp_f<0x162>(v[r]));  // xor 2
#pragma unroll
  for (int r = 0; r < 8; ++r) v[r] = fmaxf(v[r], dpp_f<0x164>(v[r]));  // xor 4
#pragma unroll
  for (int r = 0; r < 8; ++r) v[r] = fmaxf(v[r], dpp_f<0x168>(v[r]));  // xor 8
}
__device__ __forceinline__ void redsum8(float v[8]) {
#pragma unroll
  for (int r = 0; r < 8; ++r) v[r] += dpp_f<0x161>(v[r]);
#pragma unroll
  for (int r = 0; r < 8; ++r) v[r] += dpp_f<0x162>(v[r]);
#pragma unroll
  for (int r = 0; r < 8; ++r) v[r] += dpp_f<0x164>(v[r]);
#pragma unroll
  for (int r = 0; r < 8; ++r) v[r] += dpp_f<0x168>(v[r]);
}

__global__ __launch_bounds__(256) void fused_attn_ln_kernel(
    const float* __restrict__ x, const float* __restrict__ gamma,
    const float* __restrict__ beta, float* __restrict__ out)
{
  const int b    = blockIdx.x >> 4;   // 16 q-tiles per batch
  const int qt   = blockIdx.x & 15;
  const int wave = threadIdx.x >> 5;
  const int lane = threadIdx.x & 31;
  const int hid  = lane >> 4;         // wave32 half
  const int ln16 = lane & 15;

  const float* xb = x + (size_t)b * TDIM * DDIM;

  // double-buffered KV tile in two layouts + per-wave P staging (~90 KB, WGP has 320 KB)
  __shared__ alignas(16) f16 kvR[2][KTILE][DDIM + 8];   // keys row-major (Q*K^T B-frags)
  __shared__ alignas(16) f16 kvT[2][DDIM][KPAD];        // keys D-major   (P*V  B-frags)
  __shared__ alignas(16) f16 pstg[8][16][KPAD];         // C-layout -> A-layout bridge

  // ---- Q fragments: 16 rows x D=128 as 4x A(16x32 f16) ----
  const int qbase = qt * QTILE + wave * 16;
  const float* qptr = xb + (size_t)(qbase + ln16) * DDIM;  // row M = lane%16
  v16h qa[4];
#pragma unroll
  for (int c = 0; c < 4; ++c) {
    const float* p0 = qptr + c * 32 + hid * 8;  // K = c*32 + (i/8)*16 + hid*8 + i%8
#pragma unroll
    for (int j = 0; j < 8; ++j) qa[c][j] = (f16)p0[j];
#pragma unroll
    for (int j = 0; j < 8; ++j) qa[c][8 + j] = (f16)p0[16 + j];
  }

  v8f o[8] = {};
  float mrow[8], lrow[8];
#pragma unroll
  for (int r = 0; r < 8; ++r) { mrow[r] = -1e30f; lrow[r] = 0.f; }

  // staging map: 64 rows x 4 threads, 32 elems each
  const int sr  = threadIdx.x >> 2;
  const int sc4 = (threadIdx.x & 3) * 32;

  alignas(16) float pref[32];

  // ---- prologue: stage tile 0 ----
  {
    const float* src = xb + (size_t)sr * DDIM + sc4;
#pragma unroll
    for (int j = 0; j < 8; ++j) *(float4*)&pref[j * 4] = *(const float4*)(src + j * 4);
    f16 h[32];
#pragma unroll
    for (int j = 0; j < 32; ++j) h[j] = (f16)pref[j];
#pragma unroll
    for (int j = 0; j < 4; ++j) *(v8h*)&kvR[0][sr][sc4 + j * 8] = *(v8h*)&h[j * 8];
#pragma unroll
    for (int j = 0; j < 32; ++j) kvT[0][sc4 + j][sr] = h[j];
  }
  __syncthreads();

  for (int kt = 0; kt < NKT; ++kt) {
    const int cur = kt & 1;

    // ---- issue next tile's global loads early (latency hidden behind compute) ----
    if (kt + 1 < NKT) {
      const float* src = xb + (size_t)((kt + 1) * KTILE + sr) * DDIM + sc4;
#pragma unroll
      for (int j = 0; j < 8; ++j) *(float4*)&pref[j * 4] = *(const float4*)(src + j * 4);
    }

    // ---- S = Q * K^T  (16 queries x 64 keys) ----
    v8f s[4] = {};
#pragma unroll
    for (int c = 0; c < 4; ++c) {
#pragma unroll
      for (int n = 0; n < 4; ++n) {
        v16h bf;
        const f16* p = &kvR[cur][n * 16 + ln16][c * 32 + hid * 8];
        v8h u = *(const v8h*)p, v = *(const v8h*)(p + 16);
#pragma unroll
        for (int j = 0; j < 8; ++j) { bf[j] = u[j]; bf[8 + j] = v[j]; }
        s[n] = __builtin_amdgcn_wmma_f32_16x16x32_f16(false, qa[c], false, bf, (short)0, s[n], false, false);
      }
    }

    // ---- online softmax (fp32, DPP butterflies, rows interleaved) ----
    float red[8];
#pragma unroll
    for (int r = 0; r < 8; ++r)
      red[r] = fmaxf(fmaxf(s[0][r], s[1][r]), fmaxf(s[2][r], s[3][r]));
    redmax8(red);

    float sc[8];
#pragma unroll
    for (int r = 0; r < 8; ++r) {
      float mnew = fmaxf(mrow[r], red[r]);
      sc[r] = __expf(mrow[r] - mnew);
      mrow[r] = mnew;
    }
#pragma unroll
    for (int n = 0; n < 4; ++n)
#pragma unroll
      for (int r = 0; r < 8; ++r) s[n][r] = __expf(s[n][r] - mrow[r]);

    float ps[8];
#pragma unroll
    for (int r = 0; r < 8; ++r)
      ps[r] = (s[0][r] + s[1][r]) + (s[2][r] + s[3][r]);
    redsum8(ps);
#pragma unroll
    for (int r = 0; r < 8; ++r) lrow[r] = lrow[r] * sc[r] + ps[r];
#pragma unroll
    for (int d = 0; d < 8; ++d)
#pragma unroll
      for (int r = 0; r < 8; ++r) o[d][r] *= sc[r];

    // ---- P: C-layout -> A-layout via wave-private LDS (in-order DS per wave) ----
#pragma unroll
    for (int r = 0; r < 8; ++r)
#pragma unroll
      for (int n = 0; n < 4; ++n)
        pstg[wave][hid * 8 + r][n * 16 + ln16] = (f16)s[n][r];

    v16h pa[2];
#pragma unroll
    for (int ks = 0; ks < 2; ++ks) {
      const f16* p = &pstg[wave][ln16][ks * 32 + hid * 8];
      v8h u = *(const v8h*)p, v = *(const v8h*)(p + 16);
#pragma unroll
      for (int j = 0; j < 8; ++j) { pa[ks][j] = u[j]; pa[ks][8 + j] = v[j]; }
    }

    // ---- O += P * V ----
#pragma unroll
    for (int ks = 0; ks < 2; ++ks) {
#pragma unroll
      for (int d = 0; d < 8; ++d) {
        v16h vb;
        const f16* p = &kvT[cur][d * 16 + ln16][ks * 32 + hid * 8];
        v8h u = *(const v8h*)p, v = *(const v8h*)(p + 16);
#pragma unroll
        for (int j = 0; j < 8; ++j) { vb[j] = u[j]; vb[8 + j] = v[j]; }
        o[d] = __builtin_amdgcn_wmma_f32_16x16x32_f16(false, pa[ks], false, vb, (short)0, o[d], false, false);
      }
    }

    // ---- convert + store prefetched tile into the other buffer ----
    if (kt + 1 < NKT) {
      f16 h[32];
#pragma unroll
      for (int j = 0; j < 32; ++j) h[j] = (f16)pref[j];
#pragma unroll
      for (int j = 0; j < 4; ++j) *(v8h*)&kvR[1 - cur][sr][sc4 + j * 8] = *(v8h*)&h[j * 8];
#pragma unroll
      for (int j = 0; j < 32; ++j) kvT[1 - cur][sc4 + j][sr] = h[j];
    }
    __syncthreads();
  }

  // ---- epilogue: O/l + x, LayerNorm over D, gamma/beta ----
  float g[8], be[8];
#pragma unroll
  for (int d = 0; d < 8; ++d) { g[d] = gamma[d * 16 + ln16]; be[d] = beta[d * 16 + ln16]; }

  // y folded into o
#pragma unroll
  for (int r = 0; r < 8; ++r) {
    const float inv = 1.0f / lrow[r];
    const float* xr = xb + (size_t)(qbase + hid * 8 + r) * DDIM;
#pragma unroll
    for (int d = 0; d < 8; ++d) o[d][r] = o[d][r] * inv + xr[d * 16 + ln16];
  }

  float ysum[8];
#pragma unroll
  for (int r = 0; r < 8; ++r) {
    float t = 0.f;
#pragma unroll
    for (int d = 0; d < 8; ++d) t += o[d][r];
    ysum[r] = t;
  }
  redsum8(ysum);
  float mu[8];
#pragma unroll
  for (int r = 0; r < 8; ++r) mu[r] = ysum[r] * (1.0f / DDIM);

  float vs[8];
#pragma unroll
  for (int r = 0; r < 8; ++r) {
    float t = 0.f;
#pragma unroll
    for (int d = 0; d < 8; ++d) { float dv = o[d][r] - mu[r]; t += dv * dv; }
    vs[r] = t;
  }
  redsum8(vs);

  float* outb = out + (size_t)b * TDIM * DDIM;
#pragma unroll
  for (int r = 0; r < 8; ++r) {
    const float rstd = rsqrtf(vs[r] * (1.0f / DDIM) + 1e-5f);
    float* orow = outb + (size_t)(qbase + hid * 8 + r) * DDIM;
#pragma unroll
    for (int d = 0; d < 8; ++d)
      orow[d * 16 + ln16] = (o[d][r] - mu[r]) * rstd * g[d] + be[d];
  }
}

extern "C" void kernel_launch(void* const* d_in, const int* in_sizes, int n_in,
                              void* d_out, int out_size, void* d_ws, size_t ws_size,
                              hipStream_t stream) {
  const float* x     = (const float*)d_in[0];
  const float* gamma = (const float*)d_in[1];
  const float* beta  = (const float*)d_in[2];
  float* out = (float*)d_out;
  (void)in_sizes; (void)n_in; (void)out_size; (void)d_ws; (void)ws_size;

  dim3 grid(BDIM * (TDIM / QTILE));   // 256 blocks
  dim3 block(256);                    // 8 waves (wave32)
  fused_attn_ln_kernel<<<grid, block, 0, stream>>>(x, gamma, beta, out);
}
